// AutoCorr2D_53429393162398
// MI455X (gfx1250) — matrix operand order repaired and checked by reading, not compiled, not run on hardware
//
#include <hip/hip_runtime.h>

// CDNA5 / gfx1250: wave32, v_wmma_f32_16x16x32_bf16, TDM tensor_load_to_lds.
typedef __attribute__((ext_vector_type(16))) __bf16       bf16x16;
typedef __attribute__((ext_vector_type(8)))  float        f32x8;
typedef __attribute__((ext_vector_type(4)))  float        f32x4;
typedef __attribute__((ext_vector_type(4)))  unsigned int u32x4;
typedef __attribute__((ext_vector_type(8)))  int          i32x8;
typedef __attribute__((ext_vector_type(4)))  int          i32x4;

#define B_    8
#define CIN_  256
#define H_    128
#define W_    128
#define CC_   16
#define COUT_ 64
#define HW_   (H_ * W_)          // 16384
#define NPOS  (B_ * HW_)         // 131072 pixels
#define NTILE (NPOS / 16)        // 8192 wave tiles (16 px, never cross a row: W=128)

#if defined(__has_builtin)
#  if __has_builtin(__builtin_amdgcn_tensor_load_to_lds) && \
      __has_builtin(__builtin_amdgcn_s_wait_tensorcnt)
#    define HAVE_TDM 1
#  endif
#endif
#ifndef HAVE_TDM
#  define HAVE_TDM 0
#endif

// ---------------------------------------------------------------------------
// Stage 1: f[b,h,w,c] = conv3x3(x, W_ext) + b_ext   (implicit GEMM, bf16 WMMA)
//   One wave per block so the TDM descriptor is SGPR-uniform by construction.
//   Per (kh, cin-chunk): one TDM DMA of a 32cin x 18w f32 tile into LDS
//   (row stride H*W in global, tight in LDS); serves all 3 kw taps -> 3 WMMAs.
//   tensor_dim0 = 128-wstart makes TDM zero-fill the right edge; left edge &
//   out-of-range rows are predicated to 0 at fragment build.
// ---------------------------------------------------------------------------
__global__ __launch_bounds__(32) void ac2d_conv_wmma(
    const float* __restrict__ x, const float* __restrict__ Wext,
    const float* __restrict__ bext, float* __restrict__ fbuf)
{
    __shared__ float xtile[32 * 18];                // 2304 B per 1-wave block

    const int lane = threadIdx.x & 31;
    const int hi   = lane >> 4;
    const int lo   = lane & 15;

    const int tile = blockIdx.x;                    // 0..8191
    const int p0   = tile * 16;
    const int bb   = p0 >> 14;                      // / HW_
    const int hw0  = p0 & (HW_ - 1);
    const int h    = hw0 >> 7;
    const int w0   = hw0 & 127;
    const int wstart = (w0 == 0) ? 0 : (w0 - 1);    // left edge of 18-wide window

    f32x8 acc = {};
    for (int kh = 0; kh < 3; ++kh) {
        const int  h2  = h + kh - 1;
        const bool hok = (unsigned)h2 < (unsigned)H_;
        for (int cchunk = 0; cchunk < 8; ++cchunk) {
            const int cbase = cchunk * 32;

#if HAVE_TDM
            if (hok) {
                // ---- Tensor Data Mover: 2D tile (32 x 18 f32) global -> LDS
                const unsigned long long gaddr =
                    (unsigned long long)(uintptr_t)x +
                    4ull * ((unsigned long long)(bb * CIN_ + cbase) * HW_ +
                            (unsigned long long)h2 * W_ + (unsigned)wstart);
                const unsigned int ldsa =
                    (unsigned int)(uintptr_t)&xtile[0];     // low 32b = LDS offset
                const int td0 = 128 - wstart;               // right-edge zero-fill
                u32x4 g0;
                g0[0] = 1u;                                 // count=1, user D#
                g0[1] = ldsa;                               // lds_addr
                g0[2] = (unsigned int)gaddr;                // global_addr[31:0]
                g0[3] = (unsigned int)(gaddr >> 32) | 0x80000000u; // [56:32]|type=2
                i32x8 g1;
                g1[0] = (2 << 16);                          // data_size = 4B
                g1[1] = (td0 & 0xffff) << 16;               // tensor_dim0 lo
                g1[2] = (td0 >> 16) | (32 << 16);           // dim0 hi | tensor_dim1=32
                g1[3] = (18 << 16);                         // tile_dim0 = 18
                g1[4] = 32;                                 // tile_dim1 = 32
                g1[5] = HW_;                                // tensor_dim0_stride lo
                g1[6] = 0;
                g1[7] = 0;
                const i32x4 gz4 = {0, 0, 0, 0};             // groups 2/3 unused (2D)
                const i32x8 gz8 = {0, 0, 0, 0, 0, 0, 0, 0}; // 6-arg variant extra grp
                __builtin_amdgcn_tensor_load_to_lds(g0, g1, gz4, gz4, gz8, 0);
                __builtin_amdgcn_s_wait_tensorcnt(0);
            }
#endif
            for (int kw = 0; kw < 3; ++kw) {
                const int  w2 = w0 + lo + kw - 1;           // lane's pixel column
                const bool ok = hok && ((unsigned)w2 < (unsigned)W_);
                const int  wf = kh * 3 + kw;
                bf16x16 a, bf;
                #pragma unroll
                for (int i = 0; i < 16; ++i) {              // A: lane holds M=lo
                    const int ka = (i < 8 ? i : i + 8) + 8 * hi;
                    a[i] = (__bf16)Wext[(lo * CIN_ + (cbase + ka)) * 9 + wf];
                }
                #pragma unroll
                for (int i = 0; i < 16; ++i) {              // B: lane N=lo, k=(i<8?i:i+8)+8*hi
                    const int cl = (i < 8 ? i : i + 8) + 8 * hi; // cin within chunk
#if HAVE_TDM
                    const float v = ok ? xtile[cl * 18 + (w2 - wstart)] : 0.0f;
#else
                    const float v = ok ? x[(bb * CIN_ + cbase + cl) * HW_ +
                                           h2 * W_ + w2] : 0.0f;
#endif
                    bf[i] = (__bf16)v;
                }
                acc = __builtin_amdgcn_wmma_f32_16x16x32_bf16(
                          false, a, false, bf, (short)0, acc, false, false);
            }
        }
    }

    // D: lane -> pixel (N=lo), vgpr r -> channel c = r + 8*hi.  Two 16B stores.
    const int p = p0 + lo;
    float* dst = fbuf + p * CC_ + 8 * hi;
    f32x4 s0, s1;
    #pragma unroll
    for (int r = 0; r < 4; ++r) {
        s0[r] = acc[r]     + bext[r + 8 * hi];
        s1[r] = acc[r + 4] + bext[r + 4 + 8 * hi];
    }
    *(f32x4*)(dst)     = s0;
    *(f32x4*)(dst + 4) = s1;
}

// ---------------------------------------------------------------------------
// Stage 2: out[b,o,h,w] = sum_{c,k25} Wreg[o,c,k25] * f[p,c] * f[p+off(k25),c]
//   M=COUT (4x16 tiles) x N=16 pixels, K=400 (k = k25*16 + c) in 13 chunks.
//   W_reg staged in LDS pre-swizzled into WMMA A-fragment order, so each
//   fragment is ONE contiguous 32B LDS read (2x ds_load_b128), zero conflicts.
//   Corr B-fragment built once per chunk, reused across the 4 COUT tiles.
// ---------------------------------------------------------------------------
__global__ __launch_bounds__(256) void ac2d_regress_wmma(
    const float* __restrict__ fbuf, const float* __restrict__ Wreg,
    float* __restrict__ out)
{
    __shared__ __bf16 sA[13 * 4 * 32 * 16];         // 53248 B, fragment-ordered
    const int tid = threadIdx.x;
    for (int idx = tid; idx < 13 * 4 * 32 * 16; idx += 256) {
        const int i  = idx & 15;
        const int ln = (idx >> 4) & 31;
        const int ot = (idx >> 9) & 3;
        const int j  = idx >> 11;
        const int shi = ln >> 4, slo = ln & 15;
        const int ka  = i + 16 * shi;
        const int k   = 32 * j + ka;
        const int k25 = k >> 4;
        const int c   = k & 15;
        const int o   = ot * 16 + slo;
        sA[idx] = (k25 < 25) ? (__bf16)Wreg[o * 400 + c * 25 + k25] : (__bf16)0.0f;
    }
    __syncthreads();

    const int lane = tid & 31;
    const int hi   = lane >> 4;
    const int lo   = lane & 15;

    const int tile = blockIdx.x * 8 + (tid >> 5);
    const int p0   = tile * 16;
    const int bb   = p0 >> 14;
    const int hw0  = p0 & (HW_ - 1);
    const int h    = hw0 >> 7;
    const int w0   = hw0 & 127;

    const int p = p0 + lo;                          // this lane's pixel
    const int w = w0 + lo;

    // Hoist f[p][0..15] (64B contiguous -> global_load_b128 x4)
    const float* fp = fbuf + p * CC_;
    float fv[16];
    #pragma unroll
    for (int i = 0; i < 16; ++i) fv[i] = fp[i];

    f32x8 acc[4] = {};
    for (int j = 0; j < 13; ++j) {
        // B (corr) fragment: element i -> k = 32j + 16*hi + i -> k25 = 2j+hi, c = i
        const int k25b = 2 * j + hi;
        bf16x16 bf;
        if (k25b < 25) {
            const int  di = k25b / 5 - 2, dj = k25b % 5 - 2;
            const int  h2 = h + di,       w2 = w + dj;
            const bool ok = ((unsigned)h2 < (unsigned)H_) &&
                            ((unsigned)w2 < (unsigned)W_);
            const float* np = fbuf + (p + di * W_ + dj) * CC_;
            #pragma unroll
            for (int i = 0; i < 16; ++i) {
                const float nv = ok ? np[i] : 0.0f;
                bf[i] = (__bf16)(fv[i] * nv);       // corr in f32, then narrow
            }
        } else {
            #pragma unroll
            for (int i = 0; i < 16; ++i) bf[i] = (__bf16)0.0f;
        }

        // 4 COUT tiles share the B fragment; A = one 32B contiguous LDS read
        #pragma unroll
        for (int ot = 0; ot < 4; ++ot) {
            const bf16x16 a =
                *(const bf16x16*)(sA + ((j * 4 + ot) * 32 + lane) * 16);
            acc[ot] = __builtin_amdgcn_wmma_f32_16x16x32_bf16(
                          false, a, false, bf, (short)0, acc[ot], false, false);
        }
    }

    // D: vgpr r -> cout o = ot*16 + r + 8*hi, lane -> pixel. Lane-contiguous stores.
    const int hw = h * W_ + w;
    #pragma unroll
    for (int ot = 0; ot < 4; ++ot) {
        #pragma unroll
        for (int r = 0; r < 8; ++r) {
            const int o = ot * 16 + r + 8 * hi;
            out[(bb * COUT_ + o) * HW_ + hw] = acc[ot][r];
        }
    }
}

// ---------------------------------------------------------------------------
extern "C" void kernel_launch(void* const* d_in, const int* in_sizes, int n_in,
                              void* d_out, int out_size, void* d_ws, size_t ws_size,
                              hipStream_t stream) {
    (void)in_sizes; (void)n_in; (void)out_size; (void)ws_size;
    const float* x    = (const float*)d_in[0];   // (8,256,128,128)
    const float* Wext = (const float*)d_in[1];   // (16,256,3,3)
    const float* bext = (const float*)d_in[2];   // (16,)
    const float* Wreg = (const float*)d_in[3];   // (64,16,5,5)
    float*       out  = (float*)d_out;           // (8,64,128,128)
    float*       fbuf = (float*)d_ws;            // f: NPOS*CC floats = 8.4 MB scratch

    ac2d_conv_wmma<<<dim3(NTILE), dim3(32), 0, stream>>>(x, Wext, bext, fbuf);
    ac2d_regress_wmma<<<dim3(NTILE / 8), dim3(256), 0, stream>>>(fbuf, Wreg, out);
}
